// GConvLSTM_GAT_24094766531071
// MI455X (gfx1250) — compile-verified, matrix operand order checked by reading, not run on hardware
//
#include <hip/hip_runtime.h>
#include <hip/hip_bf16.h>
#include <math.h>

typedef __attribute__((ext_vector_type(16))) _Float16 v16h;
typedef __attribute__((ext_vector_type(8)))  _Float16 v8h;
typedef __attribute__((ext_vector_type(8)))  float    v8f;

#define DD 128  // feature dim (DIN == DOUT == 128)

// ---------------------------------------------------------------------------
// One-time converts: fp32 -> f16 activations; fp32 -> f16 transposed weights
// ---------------------------------------------------------------------------
__global__ void cvt_f16(const float* __restrict__ in, _Float16* __restrict__ out, int n) {
    int idx = blockIdx.x * blockDim.x + threadIdx.x;
    if (idx < n) out[idx] = (_Float16)in[idx];
}

// 4 matrices of [128,128]: Wt[mat][c][k] = (f16) W[mat][k][c]
__global__ void cvt_w_transpose(const float* __restrict__ W, _Float16* __restrict__ Wt) {
    int idx = blockIdx.x * blockDim.x + threadIdx.x;   // 4*128*128 threads
    int mat = idx >> 14, k = (idx >> 7) & 127, c = idx & 127;
    Wt[(size_t)mat * DD * DD + (size_t)c * DD + k] = (_Float16)W[idx];
}

// ---------------------------------------------------------------------------
// Y = act @ W + b  via v_wmma_f32_16x16x32_f16, f16 operands pre-staged.
// block = 256 threads = 8 waves; block covers 16 rows x 128 cols,
// wave w -> one 16x16 tile (cols [16w,16w+16)).  K = 128 = 4 WMMA steps.
// A frag: lane = row (lane&15), kbase = (lane>>4)*8; elems 0-7 -> K+0..7,
//         elems 8-15 -> K+16..23.  B frag symmetric over columns of Wt.
// ---------------------------------------------------------------------------
__global__ __launch_bounds__(256)
void wmma_gemm_bias(const _Float16* __restrict__ actH,  // [nrows,128] f16
                    const _Float16* __restrict__ WtH,   // [128,128] f16, TRANSPOSED: [c][k]
                    const float*    __restrict__ bias,  // [128]
                    float*          __restrict__ out,   // [nrows,128]
                    int nrows) {
    const int lane    = threadIdx.x & 31;
    const int wave    = threadIdx.x >> 5;
    const int rowBase = blockIdx.x * 16;
    if (rowBase + 16 > nrows) return;   // full tiles only (tail kernel covers rest)

    const int r   = lane & 15;
    const int kb  = (lane >> 4) * 8;                 // 0 or 8
    const int col = wave * 16 + r;

    const _Float16* __restrict__ arow = actH + (size_t)(rowBase + r) * DD;
    const _Float16* __restrict__ brow = WtH  + (size_t)col * DD;

    v8f c = {};
    #pragma unroll
    for (int kk = 0; kk < DD; kk += 32) {
        v8h alo = *(const v8h*)(arow + kk + kb);
        v8h ahi = *(const v8h*)(arow + kk + kb + 16);
        v8h blo = *(const v8h*)(brow + kk + kb);
        v8h bhi = *(const v8h*)(brow + kk + kb + 16);
        v16h a = __builtin_shufflevector(alo, ahi, 0,1,2,3,4,5,6,7,8,9,10,11,12,13,14,15);
        v16h b = __builtin_shufflevector(blo, bhi, 0,1,2,3,4,5,6,7,8,9,10,11,12,13,14,15);
        c = __builtin_amdgcn_wmma_f32_16x16x32_f16(false, a, false, b,
                                                   (short)0, c, false, false);
    }

    const int rowOff = (lane >> 4) * 8;              // D: VGPR v -> M = v + 8*(lane/16)
    const float bv = bias[col];
    #pragma unroll
    for (int v = 0; v < 8; ++v)
        out[(size_t)(rowBase + rowOff + v) * DD + col] = c[v] + bv;
}

// Scalar tail for nrows % 16 (not hit for N=50000; kept for robustness).
__global__ void gemm_tail(const float* __restrict__ act, const float* __restrict__ W,
                          const float* __restrict__ bias, float* __restrict__ out,
                          int rowStart, int nrows) {
    int idx = blockIdx.x * blockDim.x + threadIdx.x;
    int r = rowStart + (idx >> 7), c = idx & 127;
    if (r >= nrows) return;
    float s = bias[c];
    for (int k = 0; k < DD; ++k) s += act[(size_t)r * DD + k] * W[(size_t)k * DD + c];
    out[(size_t)r * DD + c] = s;
}

// ---------------------------------------------------------------------------
// Per-conv init: accTmp[N*128]=0, s[N]=0, mEnc[N]=0 (encoded -inf)
// ---------------------------------------------------------------------------
__global__ void conv_init(float* __restrict__ accTmp, float* __restrict__ sSum,
                          unsigned* __restrict__ mEnc, int n /* = N*128 */) {
    int idx = blockIdx.x * blockDim.x + threadIdx.x;
    if (idx >= n) return;
    accTmp[idx] = 0.f;
    if (idx < (n >> 7)) { sSum[idx] = 0.f; mEnc[idx] = 0u; }
}

__device__ __forceinline__ unsigned enc_f32(float f) {
    unsigned b = __float_as_uint(f);
    return (b & 0x80000000u) ? ~b : (b | 0x80000000u);
}
__device__ __forceinline__ float dec_f32(unsigned k) {
    return __uint_as_float((k & 0x80000000u) ? (k & 0x7FFFFFFFu) : ~k);
}

// ---------------------------------------------------------------------------
// Edge pass 1: e = att . leaky_relu(xl[src] + xr[dst]); atomicMax m[dst]
// one wave32 per edge, float4 per lane (512B/row per wave)
// ---------------------------------------------------------------------------
__global__ __launch_bounds__(256)
void edge_score(const float* __restrict__ xl, const float* __restrict__ xr,
                const float* __restrict__ att,
                const long long* __restrict__ srcI, const long long* __restrict__ dstI,
                int nEdges, int nNodes,
                float* __restrict__ eOut, unsigned* __restrict__ mEnc) {
    int edge = blockIdx.x * 8 + (threadIdx.x >> 5);
    if (edge >= nEdges + nNodes) return;
    int lane = threadIdx.x & 31;
    int s, d;
    if (edge < nEdges) { s = (int)srcI[edge]; d = (int)dstI[edge]; }
    else               { s = d = edge - nEdges; }               // self loop

    float4 a = ((const float4*)(xl + (size_t)s * DD))[lane];
    float4 b = ((const float4*)(xr + (size_t)d * DD))[lane];
    float4 t = ((const float4*)att)[lane];
    float acc = 0.f, z;
    z = a.x + b.x; z = z > 0.f ? z : 0.2f * z; acc += t.x * z;
    z = a.y + b.y; z = z > 0.f ? z : 0.2f * z; acc += t.y * z;
    z = a.z + b.z; z = z > 0.f ? z : 0.2f * z; acc += t.z * z;
    z = a.w + b.w; z = z > 0.f ? z : 0.2f * z; acc += t.w * z;
    #pragma unroll
    for (int off = 16; off; off >>= 1) acc += __shfl_xor(acc, off, 32);
    if (lane == 0) {
        eOut[edge] = acc;
        atomicMax(mEnc + d, enc_f32(acc));
    }
}

// ---------------------------------------------------------------------------
// Edge pass 2: ex = exp(e - m[dst]); s[dst] += ex; acc[dst] += ex * xl[src]
// (division by s deferred to the per-node normalize pass)
// ---------------------------------------------------------------------------
__global__ __launch_bounds__(256)
void edge_aggregate(const float* __restrict__ xl,
                    const long long* __restrict__ srcI, const long long* __restrict__ dstI,
                    int nEdges, int nNodes,
                    const float* __restrict__ eVals, const unsigned* __restrict__ mEnc,
                    float* __restrict__ sSum, float* __restrict__ acc) {
    int edge = blockIdx.x * 8 + (threadIdx.x >> 5);
    if (edge >= nEdges + nNodes) return;
    int lane = threadIdx.x & 31;
    int s, d;
    if (edge < nEdges) { s = (int)srcI[edge]; d = (int)dstI[edge]; }
    else               { s = d = edge - nEdges; }

    float ex = __expf(eVals[edge] - dec_f32(mEnc[d]));
    if (lane == 0) atomicAdd(sSum + d, ex);
    float4 a = ((const float4*)(xl + (size_t)s * DD))[lane];
    float* row = acc + (size_t)d * DD + lane * 4;
    atomicAdd(row + 0, ex * a.x);
    atomicAdd(row + 1, ex * a.y);
    atomicAdd(row + 2, ex * a.z);
    atomicAdd(row + 3, ex * a.w);
}

// ---------------------------------------------------------------------------
// Per-node normalize: gate (+)= acc/s + bias   (accumulate=0: assign; 1: add)
// ---------------------------------------------------------------------------
__global__ void normalize_add(const float* __restrict__ acc, const float* __restrict__ sSum,
                              const float* __restrict__ bias, float* __restrict__ gate,
                              int accumulate, int n) {
    int idx = blockIdx.x * blockDim.x + threadIdx.x;
    if (idx >= n) return;
    float v = acc[idx] / sSum[idx >> 7] + bias[idx & 127];
    gate[idx] = accumulate ? (gate[idx] + v) : v;
}

// ---------------------------------------------------------------------------
// Final LSTM-style gates -> Hn, Cn
// ---------------------------------------------------------------------------
__device__ __forceinline__ float sigm(float x) { return 1.f / (1.f + __expf(-x)); }

__global__ void gate_combine(const float* __restrict__ g0, const float* __restrict__ g1,
                             const float* __restrict__ g2, const float* __restrict__ g3,
                             const float* __restrict__ C,  const float* __restrict__ w_c,
                             const float* __restrict__ b_g,
                             float* __restrict__ Hn, float* __restrict__ Cn, int n) {
    int idx = blockIdx.x * blockDim.x + threadIdx.x;
    if (idx >= n) return;
    int c = idx & 127;
    float cp = C[idx];
    float I  = sigm(g0[idx] + w_c[0 * DD + c] * cp + b_g[0 * DD + c]);
    float F  = sigm(g1[idx] + w_c[1 * DD + c] * cp + b_g[1 * DD + c]);
    float T  = tanhf(g2[idx] + b_g[2 * DD + c]);
    float cn = F * cp + I * T;
    float O  = sigm(g3[idx] + w_c[2 * DD + c] * cn + b_g[3 * DD + c]);
    Hn[idx] = O * tanhf(cn);
    Cn[idx] = cn;
}

// ---------------------------------------------------------------------------
// Host orchestration
// ---------------------------------------------------------------------------
extern "C" void kernel_launch(void* const* d_in, const int* in_sizes, int n_in,
                              void* d_out, int out_size, void* d_ws, size_t ws_size,
                              hipStream_t stream) {
    const float* X   = (const float*)d_in[0];
    const float* H   = (const float*)d_in[1];
    const float* C   = (const float*)d_in[2];
    const long long* EI = (const long long*)d_in[3];
    const float* Wx_l = (const float*)d_in[4];
    const float* bx_l = (const float*)d_in[5];
    const float* Wx_r = (const float*)d_in[6];
    const float* bx_r = (const float*)d_in[7];
    const float* attx = (const float*)d_in[8];
    const float* biasx= (const float*)d_in[9];
    const float* Wh_l = (const float*)d_in[10];
    const float* bh_l = (const float*)d_in[11];
    const float* Wh_r = (const float*)d_in[12];
    const float* bh_r = (const float*)d_in[13];
    const float* atth = (const float*)d_in[14];
    const float* biash= (const float*)d_in[15];
    const float* w_c  = (const float*)d_in[16];
    const float* b_g  = (const float*)d_in[17];

    const int N = in_sizes[0] / DD;
    const int E = in_sizes[3] / 2;
    const int Etot = E + N;
    const size_t NF = (size_t)N * DD;
    const size_t WSZ = (size_t)4 * DD * DD;          // elems in one [4,128,128] group

    const long long* srcI = EI;        // edge_index[0]
    const long long* dstI = EI + E;    // edge_index[1]

    // ---- workspace layout ----
    float* ws      = (float*)d_ws;
    float* xl      = ws;                    // NF f32
    float* xr      = xl + NF;               // NF
    float* accTmp  = xr + NF;               // NF
    float* gate[4] = { accTmp + NF, accTmp + 2 * NF, accTmp + 3 * NF, accTmp + 4 * NF };
    float* sSum    = accTmp + 5 * NF;       // N
    unsigned* mEnc = (unsigned*)(sSum + N); // N
    float* eVals   = (float*)(mEnc + N);    // Etot
    _Float16* Xh   = (_Float16*)(eVals + Etot);      // NF f16
    _Float16* Hh   = Xh + NF;                        // NF f16
    _Float16* WtX_l = Hh + NF;                       // 4*128*128 f16 each
    _Float16* WtX_r = WtX_l + WSZ;
    _Float16* WtH_l = WtX_r + WSZ;
    _Float16* WtH_r = WtH_l + WSZ;

    const int gemmGrid = N / 16;
    const int gemmRem  = N - gemmGrid * 16;
    const int edgeGrid = (Etot + 7) / 8;
    const int nodeGrid = (int)((NF + 255) / 256);
    const int wGrid    = (int)((WSZ + 255) / 256);

    // ---- one-time precision/layout staging ----
    cvt_f16<<<nodeGrid, 256, 0, stream>>>(X, Xh, (int)NF);
    cvt_f16<<<nodeGrid, 256, 0, stream>>>(H, Hh, (int)NF);
    cvt_w_transpose<<<wGrid, 256, 0, stream>>>(Wx_l, WtX_l);
    cvt_w_transpose<<<wGrid, 256, 0, stream>>>(Wx_r, WtX_r);
    cvt_w_transpose<<<wGrid, 256, 0, stream>>>(Wh_l, WtH_l);
    cvt_w_transpose<<<wGrid, 256, 0, stream>>>(Wh_r, WtH_r);

    for (int conv = 0; conv < 8; ++conv) {
        const int k     = conv & 3;
        const int fromH = conv >> 2;                 // 0: X-convs, 1: H-convs
        const _Float16* actH = fromH ? Hh : Xh;
        const _Float16* Wtl  = (fromH ? WtH_l : WtX_l) + (size_t)k * DD * DD;
        const _Float16* Wtr  = (fromH ? WtH_r : WtX_r) + (size_t)k * DD * DD;
        const float* bl   = (fromH ? bh_l : bx_l) + (size_t)k * DD;
        const float* br   = (fromH ? bh_r : bx_r) + (size_t)k * DD;
        const float* att  = (fromH ? atth : attx) + (size_t)k * DD;
        const float* bout = (fromH ? biash : biasx) + (size_t)k * DD;

        // xl = act @ Wl + bl ; xr = act @ Wr + br
        wmma_gemm_bias<<<gemmGrid, 256, 0, stream>>>(actH, Wtl, bl, xl, N);
        wmma_gemm_bias<<<gemmGrid, 256, 0, stream>>>(actH, Wtr, br, xr, N);
        if (gemmRem) {
            const float* actF = fromH ? H : X;
            const float* WlF  = (fromH ? Wh_l : Wx_l) + (size_t)k * DD * DD;
            const float* WrF  = (fromH ? Wh_r : Wx_r) + (size_t)k * DD * DD;
            int tg = (gemmRem * DD + 255) / 256;
            gemm_tail<<<tg, 256, 0, stream>>>(actF, WlF, bl, xl, gemmGrid * 16, N);
            gemm_tail<<<tg, 256, 0, stream>>>(actF, WrF, br, xr, gemmGrid * 16, N);
        }

        conv_init<<<nodeGrid, 256, 0, stream>>>(accTmp, sSum, mEnc, (int)NF);

        edge_score<<<edgeGrid, 256, 0, stream>>>(xl, xr, att, srcI, dstI,
                                                 E, N, eVals, mEnc);
        edge_aggregate<<<edgeGrid, 256, 0, stream>>>(xl, srcI, dstI, E, N,
                                                     eVals, mEnc, sSum, accTmp);

        normalize_add<<<nodeGrid, 256, 0, stream>>>(accTmp, sSum, bout, gate[k],
                                                    /*accumulate=*/fromH, (int)NF);
    }

    float* Hn = (float*)d_out;
    float* Cn = Hn + NF;
    gate_combine<<<nodeGrid, 256, 0, stream>>>(gate[0], gate[1], gate[2], gate[3],
                                               C, w_c, b_g, Hn, Cn, (int)NF);
}
